// LSTM_notime_88235808129536
// MI455X (gfx1250) — matrix operand order repaired and checked by reading, not compiled, not run on hardware
//
#include <hip/hip_runtime.h>

#define BB 256
#define TT 512
#define II 3
#define HH 256
#define MTILE 16
#define HSTR 264        // padded bf16 elements per h_lds row (528 B, 16B-aligned, bank-rotating)
#define NTHREADS 256    // 8 waves: each owns 32 hidden cols -> 8 gate tiles
#define NWAVES 8

// Streamed W half in LDS: 8 waves * 4 streamed tiles * 8 ktiles * (32 lanes * 16 bf16)
#define FRAG_ELEMS 512                         // bf16 elems per fragment (32 lanes * 16)
#define WLDS_ELEMS (NWAVES * 4 * 8 * FRAG_ELEMS)
#define WLDS_BYTES (WLDS_ELEMS * 2)            // 262144 B
#define HLDS_BYTES (MTILE * HSTR * 2)          // 8448 B
#define SMEM_BYTES (WLDS_BYTES + HLDS_BYTES + 2 * MTILE * 4)

typedef __bf16 bf16;
typedef __attribute__((ext_vector_type(16))) bf16 v16bf;
typedef __attribute__((ext_vector_type(8)))  bf16 v8bf;
typedef __attribute__((ext_vector_type(8)))  float v8f;

__device__ __forceinline__ float sigmoidf_(float x) {
    return 1.0f / (1.0f + __expf(-x));
}
__device__ __forceinline__ float tanhf_(float x) {
    return 1.0f - 2.0f / (__expf(2.0f * x) + 1.0f);   // saturates correctly for |x| large
}

__device__ __forceinline__ v16bf join16(v8bf lo, v8bf hi) {
    v16bf a;
    #pragma unroll
    for (int e = 0; e < 8; ++e) { a[e] = lo[e]; a[e + 8] = hi[e]; }
    return a;
}

__global__ __launch_bounds__(NTHREADS, 1)
void lstm_persistent(const float* __restrict__ x,      // [B, T, I]
                     const float* __restrict__ W_ih,   // [4H, I]
                     const float* __restrict__ W_hh,   // [4H, H]
                     const float* __restrict__ b_ih,   // [4H]
                     const float* __restrict__ b_hh,   // [4H]
                     const float* __restrict__ W_lin,  // [1, H]
                     const float* __restrict__ b_lin,  // [1]
                     float* __restrict__ y)            // [B, T, 1]
{
    extern __shared__ __align__(16) char smem_raw[];
    bf16*  wlds  = (bf16*)smem_raw;                              // streamed W fragments
    bf16*  h_lds = (bf16*)(smem_raw + WLDS_BYTES);               // h broadcast tile
    float* ybuf  = (float*)(smem_raw + WLDS_BYTES + HLDS_BYTES); // [2][MTILE]

    const int tid   = threadIdx.x;
    const int lane  = tid & 31;
    const int wv    = tid >> 5;      // wave 0..7: owns hidden cols [32*wv, 32*wv+32)
    const int l16   = lane & 15;
    const int hl    = lane >> 4;     // lane half (0/1)
    const int bbase = blockIdx.x * MTILE;

    // init h=0 broadcast buffer and y accumulators
    for (int i = tid; i < MTILE * HSTR; i += NTHREADS) h_lds[i] = (bf16)0.0f;
    if (tid < 32) ybuf[tid] = 0.0f;

    // ---- Resident W half (gate tiles nt=0..3: i,f) in registers, f32 -> bf16 ----
    // Gate tile nt: q = nt>>1, gate row = 256*q + 32*wv + 16*(nt&1) + l16
    // B-frag element e  <->  K = kt*32 + hl*16 + e   (B[k][n] = W_hh[n][k])
    v16bf Bres[4][8];
    #pragma unroll
    for (int nt = 0; nt < 4; ++nt) {
        const int g = (nt >> 1) * 256 + wv * 32 + (nt & 1) * 16 + l16;
        const float* wrow = W_hh + (size_t)g * HH;
        #pragma unroll
        for (int kt = 0; kt < 8; ++kt) {
            const int kb = kt * 32 + hl * 16;
            v16bf bfr;
            #pragma unroll
            for (int e = 0; e < 16; ++e) bfr[e] = (bf16)wrow[kb + e];
            Bres[nt][kt] = bfr;
        }
    }

    // ---- Streamed W half (gate tiles nt=4..7: g,o) pre-swizzled into LDS ----
    // Fragment (wv, sn, kt) at wlds + ((wv*4+sn)*8 + kt)*512, lane-contiguous 32 B each.
    bf16* wbase = wlds + (size_t)(wv * 4) * 8 * FRAG_ELEMS;
    #pragma unroll
    for (int sn = 0; sn < 4; ++sn) {
        const int nt = 4 + sn;
        const int g = (nt >> 1) * 256 + wv * 32 + (nt & 1) * 16 + l16;
        const float* wrow = W_hh + (size_t)g * HH;
        #pragma unroll
        for (int kt = 0; kt < 8; ++kt) {
            const int kb = kt * 32 + hl * 16;
            v8bf lo, hi;
            #pragma unroll
            for (int e = 0; e < 8; ++e) { lo[e] = (bf16)wrow[kb + e]; hi[e] = (bf16)wrow[kb + 8 + e]; }
            bf16* dst = wbase + (sn * 8 + kt) * FRAG_ELEMS + lane * 16;
            *(v8bf*)dst = lo;
            *(v8bf*)(dst + 8) = hi;
        }
    }

    // ---- Per-lane constants: fused input projection + biases, linear head ----
    float wi0[8], wi1[8], wi2[8], bsum[8];
    #pragma unroll
    for (int nt = 0; nt < 8; ++nt) {
        const int g = (nt >> 1) * 256 + wv * 32 + (nt & 1) * 16 + l16;
        wi0[nt]  = W_ih[g * II + 0];
        wi1[nt]  = W_ih[g * II + 1];
        wi2[nt]  = W_ih[g * II + 2];
        bsum[nt] = b_ih[g] + b_hh[g];
    }
    const float wl0  = W_lin[wv * 32 + l16];
    const float wl1  = W_lin[wv * 32 + 16 + l16];
    const float blin = b_lin[0];

    v8f c0 = {}, c1 = {};   // cell state (rows 8*hl..8*hl+7, two 16-col subtiles)

    const bf16* hrow  = h_lds + l16 * HSTR;
    const bf16* wread = wbase + lane * 16;

    __syncthreads();

    for (int t = 0; t < TT; ++t) {
        // ---- wave 0 finalizes y for step t-1 (adds completed before last barrier) ----
        if (wv == 0 && t > 0 && lane < 16) {
            float v = ybuf[((t - 1) & 1) * MTILE + lane] + blin;
            y[(size_t)(bbase + lane) * TT + (t - 1)] = v > 0.0f ? v : 0.0f;
            ybuf[((t - 1) & 1) * MTILE + lane] = 0.0f;
        }

        // ---- accumulators seeded with xg = x @ W_ih.T + b_ih + b_hh (I=3) ----
        float xv0[8], xv1[8], xv2[8];
        #pragma unroll
        for (int r = 0; r < 8; ++r) {
            const float* xp = x + ((size_t)(bbase + hl * 8 + r) * TT + t) * II;
            xv0[r] = xp[0]; xv1[r] = xp[1]; xv2[r] = xp[2];
        }
        v8f acc[8];
        #pragma unroll
        for (int nt = 0; nt < 8; ++nt) {
            #pragma unroll
            for (int r = 0; r < 8; ++r)
                acc[nt][r] = bsum[nt] + xv0[r] * wi0[nt] + xv1[r] * wi1[nt] + xv2[r] * wi2[nt];
        }

        // ---- recurrent GEMM: software-pipelined over kt ----
        // prologue: A frag + 4 streamed B frags for kt=0
        v16bf Acur = join16(*(const v8bf*)(hrow + hl * 16),
                            *(const v8bf*)(hrow + hl * 16 + 8));
        v16bf S0 = join16(*(const v8bf*)(wread + 0 * 8 * FRAG_ELEMS),
                          *(const v8bf*)(wread + 0 * 8 * FRAG_ELEMS + 8));
        v16bf S1 = join16(*(const v8bf*)(wread + 1 * 8 * FRAG_ELEMS),
                          *(const v8bf*)(wread + 1 * 8 * FRAG_ELEMS + 8));
        v16bf S2 = join16(*(const v8bf*)(wread + 2 * 8 * FRAG_ELEMS),
                          *(const v8bf*)(wread + 2 * 8 * FRAG_ELEMS + 8));
        v16bf S3 = join16(*(const v8bf*)(wread + 3 * 8 * FRAG_ELEMS),
                          *(const v8bf*)(wread + 3 * 8 * FRAG_ELEMS + 8));

        #pragma unroll
        for (int kt = 0; kt < 8; ++kt) {
            v16bf An, N0, N1, N2, N3;
            if (kt < 7) {  // prefetch next kt while WMMAs run
                const int k0 = (kt + 1) * 32 + hl * 16;
                An = join16(*(const v8bf*)(hrow + k0), *(const v8bf*)(hrow + k0 + 8));
                const bf16* wp = wread + (kt + 1) * FRAG_ELEMS;
                N0 = join16(*(const v8bf*)(wp + 0 * 8 * FRAG_ELEMS), *(const v8bf*)(wp + 0 * 8 * FRAG_ELEMS + 8));
                N1 = join16(*(const v8bf*)(wp + 1 * 8 * FRAG_ELEMS), *(const v8bf*)(wp + 1 * 8 * FRAG_ELEMS + 8));
                N2 = join16(*(const v8bf*)(wp + 2 * 8 * FRAG_ELEMS), *(const v8bf*)(wp + 2 * 8 * FRAG_ELEMS + 8));
                N3 = join16(*(const v8bf*)(wp + 3 * 8 * FRAG_ELEMS), *(const v8bf*)(wp + 3 * 8 * FRAG_ELEMS + 8));
            }
            acc[0] = __builtin_amdgcn_wmma_f32_16x16x32_bf16(false, Acur, false, Bres[0][kt], (short)0, acc[0], false, false);
            acc[1] = __builtin_amdgcn_wmma_f32_16x16x32_bf16(false, Acur, false, Bres[1][kt], (short)0, acc[1], false, false);
            acc[2] = __builtin_amdgcn_wmma_f32_16x16x32_bf16(false, Acur, false, Bres[2][kt], (short)0, acc[2], false, false);
            acc[3] = __builtin_amdgcn_wmma_f32_16x16x32_bf16(false, Acur, false, Bres[3][kt], (short)0, acc[3], false, false);
            acc[4] = __builtin_amdgcn_wmma_f32_16x16x32_bf16(false, Acur, false, S0, (short)0, acc[4], false, false);
            acc[5] = __builtin_amdgcn_wmma_f32_16x16x32_bf16(false, Acur, false, S1, (short)0, acc[5], false, false);
            acc[6] = __builtin_amdgcn_wmma_f32_16x16x32_bf16(false, Acur, false, S2, (short)0, acc[6], false, false);
            acc[7] = __builtin_amdgcn_wmma_f32_16x16x32_bf16(false, Acur, false, S3, (short)0, acc[7], false, false);
            if (kt < 7) { Acur = An; S0 = N0; S1 = N1; S2 = N2; S3 = N3; }
        }

        // ---- wave-local LSTM cell nonlinearity (i=acc0/1, f=acc2/3, g=acc4/5, o=acc6/7) ----
        float hv0[8], hv1[8];
        #pragma unroll
        for (int r = 0; r < 8; ++r) {
            float i0 = sigmoidf_(acc[0][r]);
            float f0 = sigmoidf_(acc[2][r]);
            float g0 = tanhf_(acc[4][r]);
            float o0 = sigmoidf_(acc[6][r]);
            float cc0 = f0 * c0[r] + i0 * g0;
            c0[r] = cc0;
            hv0[r] = o0 * tanhf_(cc0);

            float i1 = sigmoidf_(acc[1][r]);
            float f1 = sigmoidf_(acc[3][r]);
            float g1 = tanhf_(acc[5][r]);
            float o1 = sigmoidf_(acc[7][r]);
            float cc1 = f1 * c1[r] + i1 * g1;
            c1[r] = cc1;
            hv1[r] = o1 * tanhf_(cc1);
        }

        __syncthreads();   // all reads of previous h complete

        // ---- publish new h (bf16) into the broadcast tile ----
        #pragma unroll
        for (int r = 0; r < 8; ++r) {
            h_lds[(hl * 8 + r) * HSTR + wv * 32 + l16]      = (bf16)hv0[r];
            h_lds[(hl * 8 + r) * HSTR + wv * 32 + 16 + l16] = (bf16)hv1[r];
        }

        // ---- fused linear head: partial dot over this wave's 32 hidden cols ----
        #pragma unroll
        for (int r = 0; r < 8; ++r) {
            float p = hv0[r] * wl0 + hv1[r] * wl1;
            p += __shfl_xor(p, 8, 32);
            p += __shfl_xor(p, 4, 32);
            p += __shfl_xor(p, 2, 32);
            p += __shfl_xor(p, 1, 32);
            if (l16 == 0) atomicAdd(&ybuf[(t & 1) * MTILE + hl * 8 + r], p);
        }

        __syncthreads();   // new h + y partials visible
    }

    // finalize last timestep
    if (wv == 0 && lane < 16) {
        float v = ybuf[((TT - 1) & 1) * MTILE + lane] + blin;
        y[(size_t)(bbase + lane) * TT + (TT - 1)] = v > 0.0f ? v : 0.0f;
    }
}

extern "C" void kernel_launch(void* const* d_in, const int* in_sizes, int n_in,
                              void* d_out, int out_size, void* d_ws, size_t ws_size,
                              hipStream_t stream) {
    const float* x     = (const float*)d_in[0];
    const float* W_ih  = (const float*)d_in[1];
    const float* W_hh  = (const float*)d_in[2];
    const float* b_ih  = (const float*)d_in[3];
    const float* b_hh  = (const float*)d_in[4];
    const float* W_lin = (const float*)d_in[5];
    const float* b_lin = (const float*)d_in[6];
    float* y = (float*)d_out;

    hipLaunchKernelGGL(lstm_persistent, dim3(BB / MTILE), dim3(NTHREADS), SMEM_BYTES, stream,
                       x, W_ih, W_hh, b_ih, b_hh, W_lin, b_lin, y);
}
